// PrunableDeepseekMoEWrapper_48653389529581
// MI455X (gfx1250) — matrix operand order repaired and checked by reading, not compiled, not run on hardware
//
#include <hip/hip_runtime.h>

typedef unsigned short u16;
typedef unsigned int   u32;
typedef unsigned long long u64;
typedef __attribute__((ext_vector_type(16))) __bf16 v16bf;
typedef __attribute__((ext_vector_type(8)))  float  v8f;

union FragBF { v16bf v; u32 u[8]; };

#define Tn   4096
#define Dh   2048
#define Ne   8
#define DFFr 1408
#define DFFs 2816
#define BM   128
#define BN   128
#define BK   32
#define LP   40   // LDS row pitch in bf16 elements (80B: 16B-aligned rows, bank-conflict-free stride)

static __device__ inline u16 f32_to_bf16_rne(float f) {
  u32 x = __float_as_uint(f);
  return (u16)((x + 0x7FFFu + ((x >> 16) & 1u)) >> 16);
}

// 32 bytes: global -> LDS, asynchronous (ASYNCcnt). IOFFSET adds to BOTH addresses.
static __device__ inline void async_copy32(u32 lds_addr, const u16* g) {
  asm volatile(
      "global_load_async_to_lds_b128 %0, %1, off\n\t"
      "global_load_async_to_lds_b128 %0, %1, off offset:16"
      :: "v"(lds_addr), "v"((u64)(size_t)g) : "memory");
}
static __device__ inline void wait_async0() {
  asm volatile("s_wait_asynccnt 0" ::: "memory");
}
static __device__ inline u32 lds_off(const u16* p) {
  return (u32)(size_t)p;   // LDS aperture lives in addr[63:32]; low 32 bits are the LDS byte address
}

__global__ void cvt_bf16_kernel(const float* __restrict__ s, u16* __restrict__ d, int n) {
  int i = blockIdx.x * blockDim.x + threadIdx.x;
  int st = gridDim.x * blockDim.x;
  for (; i < n; i += st) d[i] = f32_to_bf16_rne(s[i]);
}

__global__ void zero_cnt_kernel(int* c) {
  if ((int)threadIdx.x < Ne) c[threadIdx.x] = 0;
}

// One wave (32 lanes) per token: 8 dot products over D, softmax, top-2, append to expert lists.
__global__ __launch_bounds__(256) void router_kernel(
    const float* __restrict__ x, const float* __restrict__ gw,
    int* __restrict__ cnt, int* __restrict__ tidx, float* __restrict__ tw) {
  int wave = threadIdx.x >> 5;
  int lane = threadIdx.x & 31;
  int t = blockIdx.x * 8 + wave;
  if (t >= Tn) return;
  const float* xt = x + (size_t)t * Dh;
  float logits[Ne];
  for (int e = 0; e < Ne; ++e) {
    const float* g = gw + (size_t)e * Dh;
    float acc = 0.f;
    for (int d = lane; d < Dh; d += 32) acc += xt[d] * g[d];
    for (int off = 16; off > 0; off >>= 1) acc += __shfl_xor(acc, off, 32);
    logits[e] = acc;
  }
  if (lane == 0) {
    float mx = logits[0];
    for (int e = 1; e < Ne; ++e) mx = fmaxf(mx, logits[e]);
    float p[Ne], s = 0.f;
    for (int e = 0; e < Ne; ++e) { p[e] = __expf(logits[e] - mx); s += p[e]; }
    float inv = 1.f / s;
    for (int e = 0; e < Ne; ++e) p[e] *= inv;
    int i1 = 0;
    for (int e = 1; e < Ne; ++e) if (p[e] > p[i1]) i1 = e;
    int i2 = (i1 == 0) ? 1 : 0;
    for (int e = 0; e < Ne; ++e) if (e != i1 && p[e] > p[i2]) i2 = e;
    float w1 = p[i1], w2 = p[i2], sw = 1.f / (w1 + w2);
    w1 *= sw; w2 *= sw;
    int p1 = atomicAdd(&cnt[i1], 1);
    tidx[i1 * Tn + p1] = t; tw[i1 * Tn + p1] = w1;
    int p2 = atomicAdd(&cnt[i2], 1);
    tidx[i2 * Tn + p2] = t; tw[i2 * Tn + p2] = w2;
  }
}

// A fragment (16x32 bf16): lanes 0-15 row M=0..15 K{0-7,16-23}, lanes 16-31 K{8-15,24-31}
static __device__ inline void load_a_frag(const u16* __restrict__ s, int m0, int half, int l16, FragBF& f) {
  const u16* base = s + (m0 + l16) * LP;
#pragma unroll
  for (int j = 0; j < 8; ++j) {
    int kb = ((j >> 2) << 4) + (half << 3) + ((j & 3) << 1);
    f.u[j] = *(const u32*)(base + kb);
  }
}
// B fragment (32x16 bf16): lane gives col N; lanes 0-15 K{0-7,16-23}, lanes 16-31 K{8-15,24-31}.
static __device__ inline void load_b_frag(const u16* __restrict__ s, int n0, int half, int l16, FragBF& f) {
  const u16* base = s + (n0 + l16) * LP;
#pragma unroll
  for (int j = 0; j < 8; ++j) {
    int kb = ((j >> 2) << 4) + (half << 3) + ((j & 3) << 1);
    f.u[j] = *(const u32*)(base + kb);
  }
}

// H = silu(A @ Wg^T) * (A @ Wu^T), bf16 out. GATHER: rows come from per-expert token list.
template <bool GATHER>
__global__ __launch_bounds__(256) void moe_gate_up(
    const u16* __restrict__ X, const u16* __restrict__ Wg_all, const u16* __restrict__ Wu_all,
    u16* __restrict__ H_all, const int* __restrict__ cnt, const int* __restrict__ tidx,
    int K, int Ndim) {
  __shared__ __attribute__((aligned(16))) u16 As[2][BM * LP];
  __shared__ __attribute__((aligned(16))) u16 Bgs[2][BN * LP];
  __shared__ __attribute__((aligned(16))) u16 Bus[2][BN * LP];

  const int tid = threadIdx.x;
  const int mtile = blockIdx.y, ntile = blockIdx.x;
  int M = Tn;
  const u16* Wg = Wg_all; const u16* Wu = Wu_all;
  u16* H = H_all;
  const int* tl = nullptr;
  if (GATHER) {
    int e = blockIdx.z;
    M  = cnt[e];
    Wg = Wg_all + (size_t)e * Ndim * K;
    Wu = Wu_all + (size_t)e * Ndim * K;
    H  = H_all  + (size_t)e * (size_t)Tn * Ndim;
    tl = tidx + e * Tn;
  }
  const int mbase = mtile * BM;
  if (mbase >= M) return;               // uniform across block
  const int nbase = ntile * BN;

  // staging coords: each thread moves 32B (16 bf16) of one row per tile
  const int srow  = tid >> 1;
  const int shalf = (tid & 1) << 4;
  const int arow  = mbase + srow;
  const bool avalid = arow < M;
  const u16* aptr = X;
  if (avalid) {
    int grow = GATHER ? tl[arow] : arow;
    aptr = X + (size_t)grow * K + shalf;
  }
  const u16* bgp = Wg + (size_t)(nbase + srow) * K + shalf;
  const u16* bup = Wu + (size_t)(nbase + srow) * K + shalf;

  const int se = srow * LP + shalf;
  const u32 lA[2] = { lds_off(&As[0][se]),  lds_off(&As[1][se]) };
  const u32 lG[2] = { lds_off(&Bgs[0][se]), lds_off(&Bgs[1][se]) };
  const u32 lU[2] = { lds_off(&Bus[0][se]), lds_off(&Bus[1][se]) };

  // invalid A rows are never written by (EXEC-masked) async loads: zero both buffers once
  if (!avalid) {
    uint4 z = {0, 0, 0, 0};
    *(uint4*)&As[0][se] = z; *(uint4*)&As[0][se + 8] = z;
    *(uint4*)&As[1][se] = z; *(uint4*)&As[1][se + 8] = z;
  }
  // prologue: tile 0 into buffer 0
  if (avalid) async_copy32(lA[0], aptr);
  async_copy32(lG[0], bgp);
  async_copy32(lU[0], bup);

  const int wave = tid >> 5, lane = tid & 31;
  const int wm = wave & 1, wn = wave >> 1;
  const int half = lane >> 4, l16 = lane & 15;
  const int m0w = wm * 64, n0w = wn * 32;

  v8f accG[4][2], accU[4][2];
#pragma unroll
  for (int i = 0; i < 4; ++i)
#pragma unroll
    for (int j = 0; j < 2; ++j) { accG[i][j] = {}; accU[i][j] = {}; }

  const int nK = K / BK;
  for (int kt = 0; kt < nK; ++kt) {
    const int cur = kt & 1;
    wait_async0();        // my async writes for tile kt landed
    __syncthreads();      // everyone's landed; prior reads of the other buffer done
    if (kt + 1 < nK) {    // overlap: next tile into other buffer while computing
      const int kn = (kt + 1) * BK, nxt = cur ^ 1;
      if (avalid) async_copy32(lA[nxt], aptr + kn);
      async_copy32(lG[nxt], bgp + kn);
      async_copy32(lU[nxt], bup + kn);
    }
    FragBF a[4];
#pragma unroll
    for (int sm = 0; sm < 4; ++sm) load_a_frag(As[cur], m0w + sm * 16, half, l16, a[sm]);
#pragma unroll
    for (int sn = 0; sn < 2; ++sn) {
      FragBF bg, bu;
      load_b_frag(Bgs[cur], n0w + sn * 16, half, l16, bg);
      load_b_frag(Bus[cur], n0w + sn * 16, half, l16, bu);
#pragma unroll
      for (int sm = 0; sm < 4; ++sm) {
        accG[sm][sn] = __builtin_amdgcn_wmma_f32_16x16x32_bf16(
            false, a[sm].v, false, bg.v, (short)0, accG[sm][sn], false, false);
        accU[sm][sn] = __builtin_amdgcn_wmma_f32_16x16x32_bf16(
            false, a[sm].v, false, bu.v, (short)0, accU[sm][sn], false, false);
      }
    }
  }

  // epilogue: h = silu(g) * u -> bf16
#pragma unroll
  for (int sm = 0; sm < 4; ++sm) {
    int rows0 = mbase + m0w + sm * 16 + half * 8;
#pragma unroll
    for (int sn = 0; sn < 2; ++sn) {
      int col = nbase + n0w + sn * 16 + l16;
#pragma unroll
      for (int r = 0; r < 8; ++r) {
        int row = rows0 + r;
        if (row < M) {
          float g = accG[sm][sn][r], u = accU[sm][sn][r];
          float h = (g / (1.f + __expf(-g))) * u;
          H[(size_t)row * Ndim + col] = f32_to_bf16_rne(h);
        }
      }
    }
  }
}

// Y = H @ W^T. SCATTER: per-expert, atomic add weight*val into y[token]; else plain dense store.
template <bool SCATTER>
__global__ __launch_bounds__(256) void moe_down(
    const u16* __restrict__ H_all, const u16* __restrict__ W_all, float* __restrict__ Y,
    const int* __restrict__ cnt, const int* __restrict__ tidx, const float* __restrict__ tw,
    int K) {
  __shared__ __attribute__((aligned(16))) u16 As[2][BM * LP];
  __shared__ __attribute__((aligned(16))) u16 Bs[2][BN * LP];

  const int tid = threadIdx.x;
  const int mtile = blockIdx.y, ntile = blockIdx.x;
  int M = Tn;
  const u16* Hh = H_all; const u16* W = W_all;
  const int* tl = nullptr; const float* twl = nullptr;
  if (SCATTER) {
    int e = blockIdx.z;
    M  = cnt[e];
    Hh = H_all + (size_t)e * (size_t)Tn * K;
    W  = W_all + (size_t)e * (size_t)Dh * K;
    tl = tidx + e * Tn; twl = tw + e * Tn;
  }
  const int mbase = mtile * BM;
  if (mbase >= M) return;               // uniform
  const int nbase = ntile * BN;

  const int srow  = tid >> 1;
  const int shalf = (tid & 1) << 4;
  const int arow  = mbase + srow;
  const bool avalid = arow < M;
  const u16* aptr = avalid ? (Hh + (size_t)arow * K + shalf) : Hh;
  const u16* bp = W + (size_t)(nbase + srow) * K + shalf;

  const int se = srow * LP + shalf;
  const u32 lA[2] = { lds_off(&As[0][se]), lds_off(&As[1][se]) };
  const u32 lB[2] = { lds_off(&Bs[0][se]), lds_off(&Bs[1][se]) };

  if (!avalid) {
    uint4 z = {0, 0, 0, 0};
    *(uint4*)&As[0][se] = z; *(uint4*)&As[0][se + 8] = z;
    *(uint4*)&As[1][se] = z; *(uint4*)&As[1][se + 8] = z;
  }
  if (avalid) async_copy32(lA[0], aptr);
  async_copy32(lB[0], bp);

  const int wave = tid >> 5, lane = tid & 31;
  const int wm = wave & 1, wn = wave >> 1;
  const int half = lane >> 4, l16 = lane & 15;
  const int m0w = wm * 64, n0w = wn * 32;

  v8f acc[4][2];
#pragma unroll
  for (int i = 0; i < 4; ++i)
#pragma unroll
    for (int j = 0; j < 2; ++j) acc[i][j] = {};

  const int nK = K / BK;
  for (int kt = 0; kt < nK; ++kt) {
    const int cur = kt & 1;
    wait_async0();
    __syncthreads();
    if (kt + 1 < nK) {
      const int kn = (kt + 1) * BK, nxt = cur ^ 1;
      if (avalid) async_copy32(lA[nxt], aptr + kn);
      async_copy32(lB[nxt], bp + kn);
    }
    FragBF a[4];
#pragma unroll
    for (int sm = 0; sm < 4; ++sm) load_a_frag(As[cur], m0w + sm * 16, half, l16, a[sm]);
#pragma unroll
    for (int sn = 0; sn < 2; ++sn) {
      FragBF b;
      load_b_frag(Bs[cur], n0w + sn * 16, half, l16, b);
#pragma unroll
      for (int sm = 0; sm < 4; ++sm)
        acc[sm][sn] = __builtin_amdgcn_wmma_f32_16x16x32_bf16(
            false, a[sm].v, false, b.v, (short)0, acc[sm][sn], false, false);
    }
  }

#pragma unroll
  for (int sm = 0; sm < 4; ++sm) {
    int rows0 = mbase + m0w + sm * 16 + half * 8;
#pragma unroll
    for (int sn = 0; sn < 2; ++sn) {
      int col = nbase + n0w + sn * 16 + l16;
#pragma unroll
      for (int r = 0; r < 8; ++r) {
        int row = rows0 + r;
        if (SCATTER) {
          if (row < M) {
            int t = tl[row];
            float w = twl[row];
            atomicAdd(&Y[(size_t)t * Dh + col], w * acc[sm][sn][r]);
          }
        } else {
          Y[(size_t)row * Dh + col] = acc[sm][sn][r];
        }
      }
    }
  }
}

extern "C" void kernel_launch(void* const* d_in, const int* in_sizes, int n_in,
                              void* d_out, int out_size, void* d_ws, size_t ws_size,
                              hipStream_t stream) {
  (void)in_sizes; (void)n_in; (void)out_size; (void)ws_size;
  const float* x   = (const float*)d_in[0];
  const float* gw  = (const float*)d_in[1];
  const float* wg  = (const float*)d_in[2];
  const float* wu  = (const float*)d_in[3];
  const float* wd  = (const float*)d_in[4];
  const float* wsg = (const float*)d_in[5];
  const float* wsu = (const float*)d_in[6];
  const float* wsd = (const float*)d_in[7];
  float* y = (float*)d_out;

  char* base = (char*)d_ws;
  size_t off = 0;
  auto take = [&](size_t bytes) -> char* {
    char* p = base + off;
    off += (bytes + 255) & ~(size_t)255;
    return p;
  };
  u16* xb    = (u16*)take((size_t)Tn * Dh * 2);
  u16* wgb   = (u16*)take((size_t)Ne * DFFr * Dh * 2);
  u16* wub   = (u16*)take((size_t)Ne * DFFr * Dh * 2);
  u16* wdb   = (u16*)take((size_t)Ne * Dh * DFFr * 2);
  u16* wsgb  = (u16*)take((size_t)DFFs * Dh * 2);
  u16* wsub  = (u16*)take((size_t)DFFs * Dh * 2);
  u16* wsdb  = (u16*)take((size_t)Dh * DFFs * 2);
  u16* hr    = (u16*)take((size_t)Ne * Tn * DFFr * 2);
  u16* hs    = (u16*)take((size_t)Tn * DFFs * 2);
  int* cnt   = (int*)take(Ne * sizeof(int));
  int* tidx  = (int*)take((size_t)Ne * Tn * sizeof(int));
  float* tw  = (float*)take((size_t)Ne * Tn * sizeof(float));

  // fp32 -> bf16 once; full bf16 weight set (~173 MB) is L2-resident afterwards
  cvt_bf16_kernel<<<2048, 256, 0, stream>>>(x,   xb,   Tn * Dh);
  cvt_bf16_kernel<<<2048, 256, 0, stream>>>(wg,  wgb,  Ne * DFFr * Dh);
  cvt_bf16_kernel<<<2048, 256, 0, stream>>>(wu,  wub,  Ne * DFFr * Dh);
  cvt_bf16_kernel<<<2048, 256, 0, stream>>>(wd,  wdb,  Ne * Dh * DFFr);
  cvt_bf16_kernel<<<2048, 256, 0, stream>>>(wsg, wsgb, DFFs * Dh);
  cvt_bf16_kernel<<<2048, 256, 0, stream>>>(wsu, wsub, DFFs * Dh);
  cvt_bf16_kernel<<<2048, 256, 0, stream>>>(wsd, wsdb, Dh * DFFs);

  zero_cnt_kernel<<<1, 32, 0, stream>>>(cnt);
  router_kernel<<<Tn / 8, 256, 0, stream>>>(x, gw, cnt, tidx, tw);

  // shared expert: h_s then dense y (plain stores cover all of y)
  moe_gate_up<false><<<dim3(DFFs / BN, Tn / BM, 1), 256, 0, stream>>>(
      xb, wsgb, wsub, hs, nullptr, nullptr, Dh, DFFs);
  // routed experts: gathered h per expert
  moe_gate_up<true><<<dim3(DFFr / BN, Tn / BM, Ne), 256, 0, stream>>>(
      xb, wgb, wub, hr, cnt, tidx, Dh, DFFr);
  // shared down first (writes y), then routed down scatter-adds on top
  moe_down<false><<<dim3(Dh / BN, Tn / BM, 1), 256, 0, stream>>>(
      hs, wsdb, y, nullptr, nullptr, nullptr, DFFs);
  moe_down<true><<<dim3(Dh / BN, Tn / BM, Ne), 256, 0, stream>>>(
      hr, wdb, y, cnt, tidx, tw, DFFr);
}
